// GroupedMoEDispatcher_63668595196398
// MI455X (gfx1250) — matrix operand order repaired and checked by reading, not compiled, not run on hardware
//
#include <hip/hip_runtime.h>

// ---------------------------------------------------------------------------
// Grouped MoE FFN for MI455X (gfx1250, wave32, WMMA).
// B=16384 tokens, K=2, E=64 experts, H=512, F=1024, balanced routing
// (cap = 512 assignments/expert).
//
// Precision strategy: compensated bf16x3 ("split-f32") GEMMs on the
// v_wmma_f32_16x16x32_bf16 pipe. Each f32 operand x is split as
//   x_hi = top-16-bits(x) (bf16, truncated), x_lo = bf16(x - x_hi)
// and x*y ~= xh*yh + xh*yl + xl*yh (f32 accumulate). Per K=32 chunk this is
// 3 bf16 WMMAs instead of 8 f32 WMMAs -> ~2.7x the f32-WMMA FLOP rate with
// ~2^-15 relative error per term (f32-class for K=512/1024 reductions).
//
// Pipeline:
//   1) moe_init_kernel : zero d_out (atomic-accumulated) + expert counters
//   2) moe_route_kernel: counting-sort assignments by expert (atomics)
//   3) moe_fused_kernel: per (expert, 32-token tile):
//        LDS <- gathered X tile split into bf16 hi/lo planes
//        hid = silu(X @ W1[e])  (bf16x3 WMMA), hid hi/lo -> LDS
//        y   = hid @ W2[e]      (bf16x3 WMMA)
//        out[token] += w * y    via global_atomic_add_f32
// ---------------------------------------------------------------------------

typedef __attribute__((ext_vector_type(16))) __bf16 v16bf;
typedef __attribute__((ext_vector_type(8)))  float  v8f;

namespace {
constexpr int kB   = 16384;           // tokens
constexpr int kK   = 2;               // top_k
constexpr int kE   = 64;              // experts
constexpr int kH   = 512;             // hidden dim
constexpr int kF   = 1024;            // expert ffn dim
constexpr int kN   = kB * kK;         // 32768 assignments
constexpr int kCap = kN / kE;         // 512 per expert
constexpr int kMT  = 32;              // token rows per block tile
constexpr int kTPE = kCap / kMT;      // 16 tiles per expert
// bf16 LDS row strides, padded by 8 elems (16B) so every row stays
// 16B-aligned for ds_load_b128 A-fragment reads.
constexpr int kSX = kH + 8;           // 520
constexpr int kSH = kF + 8;           // 1032
}  // namespace

// v16bf assembled from two 16B chunks (A/B fragments are built from dwords).
union Frag {
  uint4  q[2];
  v16bf  v;
};

// pack {hi16(b), hi16(a)} into one dword (a -> low half) with v_perm_b32
__device__ __forceinline__ unsigned pk_hi(float a, float b) {
  return __builtin_amdgcn_perm(__float_as_uint(b), __float_as_uint(a),
                               0x07060302u);
}

// split 4 consecutive f32 into packed bf16-hi (2 dwords) and bf16-lo (2 dwords)
__device__ __forceinline__ void split4(float4 f, unsigned* h, unsigned* l) {
  unsigned u0 = __float_as_uint(f.x) & 0xffff0000u;
  unsigned u1 = __float_as_uint(f.y) & 0xffff0000u;
  unsigned u2 = __float_as_uint(f.z) & 0xffff0000u;
  unsigned u3 = __float_as_uint(f.w) & 0xffff0000u;
  h[0] = pk_hi(f.x, f.y);
  h[1] = pk_hi(f.z, f.w);
  float l0 = f.x - __uint_as_float(u0);
  float l1 = f.y - __uint_as_float(u1);
  float l2 = f.z - __uint_as_float(u2);
  float l3 = f.w - __uint_as_float(u3);
  l[0] = pk_hi(l0, l1);
  l[1] = pk_hi(l2, l3);
}

// -------------------------- init: zero out + counters ----------------------
__global__ void moe_init_kernel(float* __restrict__ out, int* __restrict__ counts,
                                int out_n4) {
  int i = blockIdx.x * blockDim.x + threadIdx.x;
  if (i < out_n4) {
    ((float4*)out)[i] = make_float4(0.f, 0.f, 0.f, 0.f);
  }
  if (i < kE) counts[i] = 0;
}

// -------------------------- route: counting sort by expert -----------------
__global__ void moe_route_kernel(const int* __restrict__ expert_ids,
                                 const float* __restrict__ expert_weights,
                                 int* __restrict__ counts,
                                 int* __restrict__ tok_sorted,
                                 float* __restrict__ w_sorted) {
  int n = blockIdx.x * blockDim.x + threadIdx.x;
  if (n < kN) {
    int e = expert_ids[n];
    int pos = atomicAdd(counts + e, 1);
    tok_sorted[e * kCap + pos] = n >> 1;   // token id = n / kK
    w_sorted[e * kCap + pos]   = expert_weights[n];
  }
}

// -------------------------- fused expert FFN -------------------------------
__global__ __launch_bounds__(256, 1)
void moe_fused_kernel(const float* __restrict__ X,
                      const float* __restrict__ W1,
                      const float* __restrict__ W2,
                      const int* __restrict__ tok_sorted,
                      const float* __restrict__ w_sorted,
                      float* __restrict__ out) {
  extern __shared__ unsigned short lds[];
  unsigned short* Xh = lds;                  // [kMT][kSX] bf16-hi of X tile
  unsigned short* Xl = Xh + kMT * kSX;       // [kMT][kSX] bf16-lo
  unsigned short* Hh = Xl + kMT * kSX;       // [kMT][kSH] bf16-hi of hid
  unsigned short* Hl = Hh + kMT * kSH;       // [kMT][kSH] bf16-lo
  int*   toks = (int*)(Hl + kMT * kSH);      // [kMT]
  float* wgts = (float*)(toks + kMT);        // [kMT]

  const int e    = blockIdx.x / kTPE;
  const int tile = blockIdx.x % kTPE;
  const int base = e * kCap + tile * kMT;

  const int tid  = threadIdx.x;
  const int wave = tid >> 5;          // 0..7
  const int lane = tid & 31;
  const int ml   = lane & 15;         // N index (B/C/D) / M index (A)
  const int k8   = (lane >> 4) << 3;  // A-fragment K sub-offset: 0 or 8
  const int mh   = (lane >> 4) << 3;  // C/D row half: 0 or 8

  if (tid < kMT) {
    toks[tid] = tok_sorted[base + tid];
    wgts[tid] = w_sorted[base + tid];
  }
  __syncthreads();

  // ---- gather X tile [kMT x kH], split into bf16 hi/lo planes in LDS ----
  for (int i = tid; i < kMT * (kH / 4); i += 256) {
    int r = i / (kH / 4);
    int c = (i % (kH / 4)) * 4;
    float4 v = *(const float4*)(X + (size_t)toks[r] * kH + c);
    unsigned h[2], l[2];
    split4(v, h, l);
    *(uint2*)(Xh + r * kSX + c) = make_uint2(h[0], h[1]);
    *(uint2*)(Xl + r * kSX + c) = make_uint2(l[0], l[1]);
  }
  __syncthreads();

  // =======================================================================
  // GEMM1: hid[32 x 1024] = silu(X @ W1[e]), bf16x3 on v_wmma 16x16x32.
  // Wave owns F-columns [wave*128, wave*128+128): 8 col tiles x 2 row tiles.
  // =======================================================================
  const float* W1e = W1 + (size_t)e * kH * kF;
  const int c1 = wave * 128;

  v8f acc1[2][8];
#pragma unroll
  for (int rt = 0; rt < 2; ++rt)
#pragma unroll
    for (int ct = 0; ct < 8; ++ct) acc1[rt][ct] = (v8f)(0.f);

  for (int kc = 0; kc < kH; kc += 32) {
    // A fragments (16-bit A 16x32 layout): lane<16 -> K {0..7,16..23},
    // lane>=16 -> K {8..15,24..31}; row = rt*16 + ml.
    Frag Ah0, Al0, Ah1, Al1;
    {
      const uint4* p = (const uint4*)(Xh + (size_t)ml * kSX + kc + k8);
      Ah0.q[0] = p[0]; Ah0.q[1] = p[2];
      const uint4* q = (const uint4*)(Xl + (size_t)ml * kSX + kc + k8);
      Al0.q[0] = q[0]; Al0.q[1] = q[2];
      const uint4* r = (const uint4*)(Xh + (size_t)(16 + ml) * kSX + kc + k8);
      Ah1.q[0] = r[0]; Ah1.q[1] = r[2];
      const uint4* s = (const uint4*)(Xl + (size_t)(16 + ml) * kSX + kc + k8);
      Al1.q[0] = s[0]; Al1.q[1] = s[2];
    }
#pragma unroll
    for (int ct = 0; ct < 8; ++ct) {
      // B fragment (32x16, lane = N col): lane reads W1[k][col+ml], K in table order
      const float* bp = W1e + (c1 + ct * 16 + ml);
      Frag Bh, Bl;
      unsigned* hq = (unsigned*)&Bh;
      unsigned* lq = (unsigned*)&Bl;
#pragma unroll
      for (int j = 0; j < 4; ++j) {
        int kb = kc + k8 + (j & 1) * 4 + (j >> 1) * 16;
        float4 f = make_float4(bp[(size_t)kb * kF], bp[(size_t)(kb + 1) * kF],
                               bp[(size_t)(kb + 2) * kF], bp[(size_t)(kb + 3) * kF]);
        split4(f, hq + 2 * j, lq + 2 * j);
      }
      acc1[0][ct] = __builtin_amdgcn_wmma_f32_16x16x32_bf16(
          false, Ah0.v, false, Bl.v, (short)0, acc1[0][ct], false, false);
      acc1[0][ct] = __builtin_amdgcn_wmma_f32_16x16x32_bf16(
          false, Al0.v, false, Bh.v, (short)0, acc1[0][ct], false, false);
      acc1[0][ct] = __builtin_amdgcn_wmma_f32_16x16x32_bf16(
          false, Ah0.v, false, Bh.v, (short)0, acc1[0][ct], false, false);
      acc1[1][ct] = __builtin_amdgcn_wmma_f32_16x16x32_bf16(
          false, Ah1.v, false, Bl.v, (short)0, acc1[1][ct], false, false);
      acc1[1][ct] = __builtin_amdgcn_wmma_f32_16x16x32_bf16(
          false, Al1.v, false, Bh.v, (short)0, acc1[1][ct], false, false);
      acc1[1][ct] = __builtin_amdgcn_wmma_f32_16x16x32_bf16(
          false, Ah1.v, false, Bh.v, (short)0, acc1[1][ct], false, false);
    }
  }

  // ---- SiLU, split hid into bf16 hi/lo planes in LDS ----
#pragma unroll
  for (int rt = 0; rt < 2; ++rt) {
#pragma unroll
    for (int ct = 0; ct < 8; ++ct) {
#pragma unroll
      for (int r = 0; r < 8; ++r) {
        float v = acc1[rt][ct][r];
        float s = v * (1.f / (1.f + __expf(-v)));
        int row = rt * 16 + mh + r;
        int col = c1 + ct * 16 + ml;
        unsigned hu = __float_as_uint(s) & 0xffff0000u;
        float lo = s - __uint_as_float(hu);
        Hh[row * kSH + col] = (unsigned short)(hu >> 16);
        Hl[row * kSH + col] = (unsigned short)(__float_as_uint(lo) >> 16);
      }
    }
  }
  __syncthreads();

  // =======================================================================
  // GEMM2: y[32 x 512] = hid @ W2[e], bf16x3.
  // Wave owns H-columns [wave*64, wave*64+64): 4 col tiles x 2 row tiles.
  // =======================================================================
  const float* W2e = W2 + (size_t)e * kF * kH;
  const int c2 = wave * 64;

  v8f acc2[2][4];
#pragma unroll
  for (int rt = 0; rt < 2; ++rt)
#pragma unroll
    for (int ct = 0; ct < 4; ++ct) acc2[rt][ct] = (v8f)(0.f);

  for (int kc = 0; kc < kF; kc += 32) {
    Frag Ah0, Al0, Ah1, Al1;
    {
      const uint4* p = (const uint4*)(Hh + (size_t)ml * kSH + kc + k8);
      Ah0.q[0] = p[0]; Ah0.q[1] = p[2];
      const uint4* q = (const uint4*)(Hl + (size_t)ml * kSH + kc + k8);
      Al0.q[0] = q[0]; Al0.q[1] = q[2];
      const uint4* r = (const uint4*)(Hh + (size_t)(16 + ml) * kSH + kc + k8);
      Ah1.q[0] = r[0]; Ah1.q[1] = r[2];
      const uint4* s = (const uint4*)(Hl + (size_t)(16 + ml) * kSH + kc + k8);
      Al1.q[0] = s[0]; Al1.q[1] = s[2];
    }
#pragma unroll
    for (int ct = 0; ct < 4; ++ct) {
      const float* bp = W2e + (c2 + ct * 16 + ml);
      Frag Bh, Bl;
      unsigned* hq = (unsigned*)&Bh;
      unsigned* lq = (unsigned*)&Bl;
#pragma unroll
      for (int j = 0; j < 4; ++j) {
        int kb = kc + k8 + (j & 1) * 4 + (j >> 1) * 16;
        float4 f = make_float4(bp[(size_t)kb * kH], bp[(size_t)(kb + 1) * kH],
                               bp[(size_t)(kb + 2) * kH], bp[(size_t)(kb + 3) * kH]);
        split4(f, hq + 2 * j, lq + 2 * j);
      }
      acc2[0][ct] = __builtin_amdgcn_wmma_f32_16x16x32_bf16(
          false, Ah0.v, false, Bl.v, (short)0, acc2[0][ct], false, false);
      acc2[0][ct] = __builtin_amdgcn_wmma_f32_16x16x32_bf16(
          false, Al0.v, false, Bh.v, (short)0, acc2[0][ct], false, false);
      acc2[0][ct] = __builtin_amdgcn_wmma_f32_16x16x32_bf16(
          false, Ah0.v, false, Bh.v, (short)0, acc2[0][ct], false, false);
      acc2[1][ct] = __builtin_amdgcn_wmma_f32_16x16x32_bf16(
          false, Ah1.v, false, Bl.v, (short)0, acc2[1][ct], false, false);
      acc2[1][ct] = __builtin_amdgcn_wmma_f32_16x16x32_bf16(
          false, Al1.v, false, Bh.v, (short)0, acc2[1][ct], false, false);
      acc2[1][ct] = __builtin_amdgcn_wmma_f32_16x16x32_bf16(
          false, Ah1.v, false, Bh.v, (short)0, acc2[1][ct], false, false);
    }
  }

  // ---- weighted scatter-combine: out[token] += w * y ----
#pragma unroll
  for (int rt = 0; rt < 2; ++rt) {
#pragma unroll
    for (int ct = 0; ct < 4; ++ct) {
#pragma unroll
      for (int r = 0; r < 8; ++r) {
        int m = rt * 16 + mh + r;
        int col = c2 + ct * 16 + ml;
        atomicAdd(out + (size_t)toks[m] * kH + col, acc2[rt][ct][r] * wgts[m]);
      }
    }
  }
}

// ---------------------------------------------------------------------------
extern "C" void kernel_launch(void* const* d_in, const int* in_sizes, int n_in,
                              void* d_out, int out_size, void* d_ws, size_t ws_size,
                              hipStream_t stream) {
  const float* X  = (const float*)d_in[0];   // hidden_states [B,H] f32
  const float* EW = (const float*)d_in[1];   // expert_weights [B,K] f32
  const int*   EI = (const int*)d_in[2];     // expert_ids [B,K] i32
  const float* W1 = (const float*)d_in[3];   // [E,H,F] f32
  const float* W2 = (const float*)d_in[4];   // [E,F,H] f32
  float* out = (float*)d_out;                // [B,H] f32

  // workspace layout: counts[64] | tok_sorted[32768] | w_sorted[32768]
  int*   counts     = (int*)d_ws;
  int*   tok_sorted = counts + kE;
  float* w_sorted   = (float*)(tok_sorted + kN);

  const int out_n4 = (kB * kH) / 4;
  moe_init_kernel<<<(out_n4 + 255) / 256, 256, 0, stream>>>(out, counts, out_n4);
  moe_route_kernel<<<(kN + 255) / 256, 256, 0, stream>>>(EI, EW, counts,
                                                         tok_sorted, w_sorted);

  const size_t lds_bytes =
      (size_t)(2 * kMT * kSX + 2 * kMT * kSH) * sizeof(unsigned short) +
      kMT * sizeof(int) + kMT * sizeof(float);
  moe_fused_kernel<<<kE * kTPE, 256, lds_bytes, stream>>>(
      X, W1, W2, tok_sorted, w_sorted, out);
}
